// GNN_drug_19507741458745
// MI455X (gfx1250) — compile-verified, matrix operand order checked by reading, not run on hardware
//
#include <hip/hip_runtime.h>
#include <hip/hip_bf16.h>
#include <math.h>

// ---------------------------------------------------------------------------
// GAT-based GNN forward for MI455X (gfx1250, wave32).
// Dense node GEMMs -> f16 WMMA (v_wmma_f32_16x16x32_f16, f32 accumulate).
// Edge softmax/scatter -> f32 atomics (stay resident in 192MB L2).
// ---------------------------------------------------------------------------

#define N_NODES  50000
#define N_EDGES  400000
#define E_TOT    (N_EDGES + N_NODES)   // self-loops appended
#define N_GRAPHS 1024
#define HID      312
#define F_IN     78
#define HFEAT    (2 * HID)             // 624 = HEADS*HID

typedef __attribute__((ext_vector_type(16))) _Float16 v16h;
typedef __attribute__((ext_vector_type(8)))  _Float16 v8h;
typedef __attribute__((ext_vector_type(2)))  _Float16 v2h;
typedef __attribute__((ext_vector_type(8)))  float    v8f;
typedef __attribute__((ext_vector_type(4)))  float    v4f;

#define TM  128          // block tile M
#define TN  128          // block tile N
#define LDK 40           // padded LDS row stride in halves (80B, 16B-aligned rows)

// Build one 16x32 f16 fragment from an LDS row (ISA 16-bit A/B layout:
// half t <-> K = kbase + t + (t>=8?8:0); pairs are contiguous -> b128 loads).
__device__ __forceinline__ v16h load_frag(const _Float16* __restrict__ rp, int kbase)
{
  v16h f;
#pragma unroll
  for (int p = 0; p < 8; ++p) {
    const int kk = kbase + 2 * p + (p >= 4 ? 8 : 0);
    v2h t = *(const v2h*)(rp + kk);
    f[2 * p]     = t[0];
    f[2 * p + 1] = t[1];
  }
  return f;
}

// ---------------------------------------------------------------------------
// Generic f32-in/f32-out GEMM, f16 WMMA compute, optional bias + relu.
// C[M,N] = A[M,K] * B[K,N] (+ bias[N]) (relu?)
// Block: 256 threads = 8 waves; tile 128(M) x 128(N), K step 32.
// Wave w: rows [(w&3)*32,+32), cols [(w>>2)*64,+64) -> 2 A-frags x 4 B-frags,
// all staged to registers first, then 8 back-to-back WMMAs.
// ---------------------------------------------------------------------------
__global__ __launch_bounds__(256)
void gemm_wmma_f16(const float* __restrict__ A, const float* __restrict__ B,
                   float* __restrict__ C, int M, int N, int K,
                   const float* __restrict__ bias, int relu)
{
  __shared__ __align__(16) _Float16 As[TM * LDK];   // [row][k]
  __shared__ __align__(16) _Float16 Bs[TN * LDK];   // [col][k] (transposed)

  const int tid  = threadIdx.x;
  const int lane = tid & 31;
  const int wave = tid >> 5;
  const int wm   = wave & 3;          // M sub-block (32 rows)
  const int wnh  = wave >> 2;         // N half (64 cols)
  const int half = lane >> 4;
  const int rc   = lane & 15;         // row (A) / col (B,D)
  const int kbase = half * 8;

  const int n0 = blockIdx.x * TN;
  const int m0 = blockIdx.y * TM;

  v8f acc[8] = {v8f{}, v8f{}, v8f{}, v8f{}, v8f{}, v8f{}, v8f{}, v8f{}};

  // A staging map: each thread: row aRow (0..127), 16 consecutive k
  const int aRow = tid >> 1;
  const int aKc  = (tid & 1) * 16;
  // B staging fast map: 2 k-rows x 8 n per thread
  const int bKp  = (tid >> 4) * 2;    // 0,2,..,30
  const int bNg  = (tid & 15) * 8;    // 0..120
  // B staging guarded map: 1 k-row x 16 n per thread
  const int gK   = tid >> 3;          // 0..31
  const int gNc  = (tid & 7) * 16;

  const bool mFull = (m0 + TM <= M);
  const bool nFull = (n0 + TN <= N);
  const bool aVec  = ((K & 3) == 0);

  for (int k0 = 0; k0 < K; k0 += 32) {
    const bool kFull = (k0 + 32 <= K);

    // ---------------- stage A tile ----------------
    if (mFull && kFull) {
      const float* ap = A + (size_t)(m0 + aRow) * K + (k0 + aKc);
      _Float16 t[16];
      if (aVec) {
#pragma unroll
        for (int q = 0; q < 16; q += 4) {
          v4f x = *(const v4f*)(ap + q);
          t[q] = (_Float16)x[0]; t[q+1] = (_Float16)x[1];
          t[q+2] = (_Float16)x[2]; t[q+3] = (_Float16)x[3];
        }
      } else {
#pragma unroll
        for (int q = 0; q < 16; ++q) t[q] = (_Float16)ap[q];
      }
      v8h h0, h1;
#pragma unroll
      for (int q = 0; q < 8; ++q) { h0[q] = t[q]; h1[q] = t[q + 8]; }
      *(v8h*)&As[aRow * LDK + aKc]     = h0;
      *(v8h*)&As[aRow * LDK + aKc + 8] = h1;
    } else {
      // clamped addressing, two phases: pipeline all loads, then convert
      const int gr  = m0 + aRow;
      const int grc = (gr < M) ? gr : (M - 1);
      const float* ap = A + (size_t)grc * K;
      const bool rowOk = (gr < M);
      float vbuf[16];
#pragma unroll
      for (int q = 0; q < 16; ++q) {
        const int gk  = k0 + aKc + q;
        const int gkc = (gk < K) ? gk : (K - 1);
        vbuf[q] = ap[gkc];
      }
#pragma unroll
      for (int q = 0; q < 16; ++q) {
        const bool ok = rowOk && (k0 + aKc + q < K);
        As[aRow * LDK + aKc + q] = ok ? (_Float16)vbuf[q] : (_Float16)0.0f;
      }
    }

    // ---------------- stage B tile (transposed) ----------------
    if (nFull && kFull) {
      const float* bp0 = B + (size_t)(k0 + bKp) * N + (n0 + bNg);
      const float* bp1 = bp0 + N;
      float r0[8], r1[8];
      // N is a multiple of 4 for every GEMM in this net -> v4f loads
#pragma unroll
      for (int q = 0; q < 8; q += 4) {
        v4f x0 = *(const v4f*)(bp0 + q);
        v4f x1 = *(const v4f*)(bp1 + q);
        r0[q] = x0[0]; r0[q+1] = x0[1]; r0[q+2] = x0[2]; r0[q+3] = x0[3];
        r1[q] = x1[0]; r1[q+1] = x1[1]; r1[q+2] = x1[2]; r1[q+3] = x1[3];
      }
#pragma unroll
      for (int q = 0; q < 8; ++q) {
        v2h pr; pr[0] = (_Float16)r0[q]; pr[1] = (_Float16)r1[q];
        *(v2h*)&Bs[(bNg + q) * LDK + bKp] = pr;
      }
    } else {
      const int gk  = k0 + gK;
      const int gkc = (gk < K) ? gk : (K - 1);
      const float* bp = B + (size_t)gkc * N;
      const bool kOk = (gk < K);
      float vbuf[16];
#pragma unroll
      for (int q = 0; q < 16; ++q) {
        const int gn  = n0 + gNc + q;
        const int gnc = (gn < N) ? gn : (N - 1);
        vbuf[q] = bp[gnc];
      }
#pragma unroll
      for (int q = 0; q < 16; ++q) {
        const bool ok = kOk && (n0 + gNc + q < N);
        Bs[(gNc + q) * LDK + gK] = ok ? (_Float16)vbuf[q] : (_Float16)0.0f;
      }
    }
    __syncthreads();

    // ---------------- compute: stage all 6 fragments, then 8 WMMAs ----------
    const v16h a0 = load_frag(&As[(wm * 32 + rc) * LDK], kbase);
    const v16h a1 = load_frag(&As[(wm * 32 + 16 + rc) * LDK], kbase);
    v16h bf[4];
#pragma unroll
    for (int wn = 0; wn < 4; ++wn)
      bf[wn] = load_frag(&Bs[(wnh * 64 + wn * 16 + rc) * LDK], kbase);

#pragma unroll
    for (int wn = 0; wn < 4; ++wn) {
      acc[wn]     = __builtin_amdgcn_wmma_f32_16x16x32_f16(
          false, a0, false, bf[wn], (short)0, acc[wn], false, false);
      acc[wn + 4] = __builtin_amdgcn_wmma_f32_16x16x32_f16(
          false, a1, false, bf[wn], (short)0, acc[wn + 4], false, false);
    }
    __syncthreads();
  }

  // ---- epilogue: D layout = VGPR g -> row g + half*8, col = lane&15
#pragma unroll
  for (int fr = 0; fr < 2; ++fr) {
#pragma unroll
    for (int wn = 0; wn < 4; ++wn) {
      const int col = n0 + wnh * 64 + wn * 16 + rc;
      if (col < N) {
        const float bv = bias ? bias[col] : 0.0f;
#pragma unroll
        for (int g = 0; g < 8; ++g) {
          const int row = m0 + wm * 32 + fr * 16 + half * 8 + g;
          if (row < M) {
            float v = acc[fr * 4 + wn][g] + bv;
            if (relu) v = fmaxf(v, 0.0f);
            C[(size_t)row * N + col] = v;
          }
        }
      }
    }
  }
}

// ---------------------------------------------------------------------------
// fill
// ---------------------------------------------------------------------------
__global__ void fill_kernel(float* __restrict__ p, float v, size_t n)
{
  size_t i = (size_t)blockIdx.x * blockDim.x + threadIdx.x;
  if (i < n) p[i] = v;
}

// ---------------------------------------------------------------------------
// attention scores: one wave per node.
// s_src[n,h] = sum_c H[n,h,c]*a_src[h,c];  s_dst likewise.
// ---------------------------------------------------------------------------
__global__ void score_kernel(const float* __restrict__ H,
                             const float* __restrict__ a_src,
                             const float* __restrict__ a_dst,
                             float* __restrict__ s_src, float* __restrict__ s_dst)
{
  const int wid  = (int)(((size_t)blockIdx.x * blockDim.x + threadIdx.x) >> 5);
  const int lane = threadIdx.x & 31;
  if (wid >= N_NODES) return;
  const float* row = H + (size_t)wid * HFEAT;
  float ss0 = 0.f, ss1 = 0.f, sd0 = 0.f, sd1 = 0.f;
  for (int c = lane; c < HID; c += 32) {
    const float h0 = row[c], h1 = row[c + HID];
    ss0 += h0 * a_src[c];        sd0 += h0 * a_dst[c];
    ss1 += h1 * a_src[c + HID];  sd1 += h1 * a_dst[c + HID];
  }
#pragma unroll
  for (int off = 16; off > 0; off >>= 1) {
    ss0 += __shfl_down(ss0, off, 32);
    ss1 += __shfl_down(ss1, off, 32);
    sd0 += __shfl_down(sd0, off, 32);
    sd1 += __shfl_down(sd1, off, 32);
  }
  if (lane == 0) {
    s_src[wid * 2 + 0] = ss0;  s_src[wid * 2 + 1] = ss1;
    s_dst[wid * 2 + 0] = sd0;  s_dst[wid * 2 + 1] = sd1;
  }
}

// ---------------------------------------------------------------------------
// float atomic max via CAS
// ---------------------------------------------------------------------------
__device__ inline void atomicMaxFloat(float* addr, float val)
{
  unsigned int* ua  = reinterpret_cast<unsigned int*>(addr);
  unsigned int  cur = __float_as_uint(*addr);
  while (__uint_as_float(cur) < val) {
    unsigned int prev = atomicCAS(ua, cur, __float_as_uint(val));
    if (prev == cur) break;
    cur = prev;
  }
}

// edge pass A: alpha = leaky_relu(s_src[src]+s_dst[dst], 0.2); m[dst] = max
__global__ void edge_alpha_max(const int* __restrict__ ei,
                               const float* __restrict__ s_src,
                               const float* __restrict__ s_dst,
                               float* __restrict__ alpha, float* __restrict__ mmax)
{
  const int e = blockIdx.x * blockDim.x + threadIdx.x;
  if (e >= E_TOT) return;
  const int s = (e < N_EDGES) ? ei[e] : (e - N_EDGES);
  const int t = (e < N_EDGES) ? ei[N_EDGES + e] : (e - N_EDGES);
#pragma unroll
  for (int h = 0; h < 2; ++h) {
    float a = s_src[s * 2 + h] + s_dst[t * 2 + h];
    a = (a > 0.f) ? a : 0.2f * a;
    alpha[(size_t)e * 2 + h] = a;
    atomicMaxFloat(&mmax[t * 2 + h], a);
  }
}

// edge pass B: e = exp(alpha - m[dst]) (in-place over alpha); z[dst] += e
__global__ void edge_exp(const int* __restrict__ ei,
                         const float* __restrict__ mmax,
                         float* __restrict__ alpha, float* __restrict__ zsum)
{
  const int e = blockIdx.x * blockDim.x + threadIdx.x;
  if (e >= E_TOT) return;
  const int t = (e < N_EDGES) ? ei[N_EDGES + e] : (e - N_EDGES);
#pragma unroll
  for (int h = 0; h < 2; ++h) {
    const float ev = expf(alpha[(size_t)e * 2 + h] - mmax[t * 2 + h]);
    alpha[(size_t)e * 2 + h] = ev;
    atomicAdd(&zsum[t * 2 + h], ev);
  }
}

// edge pass C: one wave per edge; acc[dst,h,c] += H[src,h,c]*coef[h]
__global__ void edge_scatter(const int* __restrict__ ei,
                             const float* __restrict__ evals,
                             const float* __restrict__ zsum,
                             const float* __restrict__ H, float* __restrict__ acc)
{
  const int wid  = (int)(((size_t)blockIdx.x * blockDim.x + threadIdx.x) >> 5);
  const int lane = threadIdx.x & 31;
  if (wid >= E_TOT) return;
  const int s = (wid < N_EDGES) ? ei[wid] : (wid - N_EDGES);
  const int t = (wid < N_EDGES) ? ei[N_EDGES + wid] : (wid - N_EDGES);
  const float c0 = evals[(size_t)wid * 2 + 0] / (zsum[t * 2 + 0] + 1e-16f);
  const float c1 = evals[(size_t)wid * 2 + 1] / (zsum[t * 2 + 1] + 1e-16f);
  const float* hrow = H + (size_t)s * HFEAT;
  float* arow = acc + (size_t)t * HFEAT;
  for (int c = lane; c < HID; c += 32) {
    atomicAdd(&arow[c],       hrow[c]       * c0);
    atomicAdd(&arow[c + HID], hrow[c + HID] * c1);
  }
}

// out[n,c] = mean over heads + bias (+ relu)
__global__ void finalize_node(const float* __restrict__ acc,
                              const float* __restrict__ bias,
                              float* __restrict__ out, int relu)
{
  const size_t i = (size_t)blockIdx.x * blockDim.x + threadIdx.x;
  if (i >= (size_t)N_NODES * HID) return;
  const size_t n = i / HID;
  const int    c = (int)(i % HID);
  float v = 0.5f * (acc[n * HFEAT + c] + acc[n * HFEAT + HID + c]) + bias[c];
  if (relu) v = fmaxf(v, 0.0f);
  out[i] = v;
}

// gated residual: z = sigmoid(xa + xb + fc2_b + mol_bias); h = z*x + (1-z)*h
__global__ void gate_kernel(const float* __restrict__ x,
                            const float* __restrict__ xa,   // x@fc1_w + fc1_b
                            const float* __restrict__ xb,   // h@fc2_w
                            const float* __restrict__ fc2_b,
                            const float* __restrict__ mol_bias,
                            float* __restrict__ h)
{
  const size_t i = (size_t)blockIdx.x * blockDim.x + threadIdx.x;
  if (i >= (size_t)N_NODES * HID) return;
  const int c = (int)(i % HID);
  const float zz = 1.0f / (1.0f + expf(-(xa[i] + xb[i] + fc2_b[c] + mol_bias[c])));
  h[i] = zz * x[i] + (1.0f - zz) * h[i];
}

// segment-sum pooling
__global__ void pool_kernel(const float* __restrict__ h, const int* __restrict__ batch,
                            float* __restrict__ pooled, float* __restrict__ cnt)
{
  const size_t i = (size_t)blockIdx.x * blockDim.x + threadIdx.x;
  if (i >= (size_t)N_NODES * HID) return;
  const size_t n = i / HID;
  const int    c = (int)(i % HID);
  const int    g = batch[n];
  atomicAdd(&pooled[(size_t)g * HID + c], h[i]);
  if (c == 0) atomicAdd(&cnt[g], 1.0f);
}

__global__ void pool_div(float* __restrict__ pooled, const float* __restrict__ cnt)
{
  const size_t i = (size_t)blockIdx.x * blockDim.x + threadIdx.x;
  if (i >= (size_t)N_GRAPHS * HID) return;
  pooled[i] /= fmaxf(cnt[i / HID], 1.0f);
}

// ---------------------------------------------------------------------------
// host orchestration
// ---------------------------------------------------------------------------
static inline int cdiv(size_t a, size_t b) { return (int)((a + b - 1) / b); }

extern "C" void kernel_launch(void* const* d_in, const int* in_sizes, int n_in,
                              void* d_out, int out_size, void* d_ws, size_t ws_size,
                              hipStream_t stream)
{
  const float* mol_x  = (const float*)d_in[0];
  const int*   ei     = (const int*)d_in[1];
  const int*   batch  = (const int*)d_in[2];
  const float* W[3]   = {(const float*)d_in[3], (const float*)d_in[7],  (const float*)d_in[11]};
  const float* AS[3]  = {(const float*)d_in[4], (const float*)d_in[8],  (const float*)d_in[12]};
  const float* AD[3]  = {(const float*)d_in[5], (const float*)d_in[9],  (const float*)d_in[13]};
  const float* BB[3]  = {(const float*)d_in[6], (const float*)d_in[10], (const float*)d_in[14]};
  const float* fc1_w  = (const float*)d_in[15];
  const float* fc1_b  = (const float*)d_in[16];
  const float* fc2_w  = (const float*)d_in[17];
  const float* fc2_b  = (const float*)d_in[18];
  const float* mbias  = (const float*)d_in[19];
  const float* g1_w   = (const float*)d_in[20];
  const float* g1_b   = (const float*)d_in[21];
  const float* g2_w   = (const float*)d_in[22];
  const float* g2_b   = (const float*)d_in[23];
  float* out = (float*)d_out;

  // workspace carve-up
  float* ws = (float*)d_ws;
  size_t o = 0;
  float* H     = ws + o; o += (size_t)N_NODES * HFEAT;   // [N,624] per-layer h = x@W
  float* acc   = ws + o; o += (size_t)N_NODES * HFEAT;   // scatter accumulator
  float* hbuf  = ws + o; o += (size_t)N_NODES * HID;     // running node state
  float* xnew  = ws + o; o += (size_t)N_NODES * HID;     // conv output
  float* ssrc  = ws + o; o += (size_t)N_NODES * 2;
  float* sdst  = ws + o; o += (size_t)N_NODES * 2;
  float* alpha = ws + o; o += (size_t)E_TOT * 2;
  float* mmax  = ws + o; o += (size_t)N_NODES * 2;
  float* zsum  = ws + o; o += (size_t)N_NODES * 2;
  float* pooled= ws + o; o += (size_t)N_GRAPHS * HID;
  float* cnt   = ws + o; o += (size_t)N_GRAPHS;
  float* t1    = ws + o; o += (size_t)N_GRAPHS * 1024;
  // xa/xb live in H's region (H is dead once acc is finalized)
  float* xa = H;
  float* xb = H + (size_t)N_NODES * HID;

  auto gemm = [&](const float* A, const float* B, float* C, int M, int N, int K,
                  const float* bias, int relu) {
    dim3 grid(cdiv(N, TN), cdiv(M, TM));
    gemm_wmma_f16<<<grid, 256, 0, stream>>>(A, B, C, M, N, K, bias, relu);
  };
  auto fill = [&](float* p, float v, size_t n) {
    fill_kernel<<<cdiv(n, 256), 256, 0, stream>>>(p, v, n);
  };

  auto gat_layer = [&](const float* X, int Kdim, int li, float* outX, int relu) {
    gemm(X, W[li], H, N_NODES, HFEAT, Kdim, nullptr, 0);
    score_kernel<<<cdiv((size_t)N_NODES * 32, 256), 256, 0, stream>>>(H, AS[li], AD[li], ssrc, sdst);
    fill(mmax, -3.0e38f, (size_t)N_NODES * 2);
    fill(zsum, 0.0f, (size_t)N_NODES * 2);
    fill(acc, 0.0f, (size_t)N_NODES * HFEAT);
    edge_alpha_max<<<cdiv(E_TOT, 256), 256, 0, stream>>>(ei, ssrc, sdst, alpha, mmax);
    edge_exp<<<cdiv(E_TOT, 256), 256, 0, stream>>>(ei, mmax, alpha, zsum);
    edge_scatter<<<cdiv((size_t)E_TOT * 32, 256), 256, 0, stream>>>(ei, alpha, zsum, H, acc);
    finalize_node<<<cdiv((size_t)N_NODES * HID, 256), 256, 0, stream>>>(acc, BB[li], outX, relu);
  };

  // layer 0: h = relu(gat(mol_x))
  gat_layer(mol_x, F_IN, 0, hbuf, 1);

  // layers 1,2 with gated residual
  for (int li = 1; li <= 2; ++li) {
    gat_layer(hbuf, HID, li, xnew, li == 1 ? 1 : 0);
    gemm(xnew, fc1_w, xa, N_NODES, HID, HID, fc1_b, 0);   // x@fc1 + fc1_b
    gemm(hbuf, fc2_w, xb, N_NODES, HID, HID, nullptr, 0); // h@fc2
    gate_kernel<<<cdiv((size_t)N_NODES * HID, 256), 256, 0, stream>>>(
        xnew, xa, xb, fc2_b, mbias, hbuf);
  }

  // global mean pool
  fill(pooled, 0.0f, (size_t)N_GRAPHS * HID);
  fill(cnt, 0.0f, (size_t)N_GRAPHS);
  pool_kernel<<<cdiv((size_t)N_NODES * HID, 256), 256, 0, stream>>>(hbuf, batch, pooled, cnt);
  pool_div<<<cdiv((size_t)N_GRAPHS * HID, 256), 256, 0, stream>>>(pooled, cnt);

  // MLP head
  gemm(pooled, g1_w, t1, N_GRAPHS, 1024, HID, g1_b, 1);
  gemm(t1, g2_w, out, N_GRAPHS, 128, 1024, g2_b, 0);
}
